// ConvOffset2d_74869869904097
// MI455X (gfx1250) — matrix operand order repaired and checked by reading, not compiled, not run on hardware
//
#include <hip/hip_runtime.h>
#include <hip/hip_bf16.h>
#include <cstdint>

typedef __attribute__((ext_vector_type(16))) _Float16 v16h;
typedef __attribute__((ext_vector_type(8)))  float    v8f;

#define N_BATCH 8
#define C_IN    128
#define C_OUT   256
#define H_IN    56
#define W_IN    56
#define P_TOT   (H_IN * W_IN)      // 3136 output positions
#define K_TAPS  9
#define K_TOT   (C_IN * K_TAPS)    // 1152 GEMM K dim
#define K_STEPS (K_TOT / 32)       // 36 wmma k-steps
#define NT      32                 // positions per workgroup tile
#define N_TILES (P_TOT / NT)       // 98 tiles per batch
#define COMBOS  (NT * 2 * K_TAPS)  // 576 (pos, dg, tap) tuples

// ---------------------------------------------------------------------------
// Kernel 0: repack weight [256][1152] f32 -> f16 fragments in WMMA A layout.
// Fragment (msub, ks) = 16x32 tile of A; per lane 16 halves (32B contiguous).
// A layout used here (contiguous k, not the ISA table order; 16-bit A 16x32):
//   M = lane%16 ;  K' = (lane>>4)*16 + t   (t = half 0..15)
// Storage: wfrag[(((msub*K_STEPS)+ks)*32 + lane)*16 + t]
// ---------------------------------------------------------------------------
__global__ __launch_bounds__(256)
void repack_weight_kernel(const float* __restrict__ w, _Float16* __restrict__ wfrag) {
    int o = blockIdx.x * 256 + threadIdx.x;          // 0 .. 16*36*512-1
    int t    = o & 15;
    int lane = (o >> 4) & 31;
    int ksm  = o >> 9;
    int ks   = ksm % K_STEPS;
    int msub = ksm / K_STEPS;
    if (msub >= 16) return;
    int M  = msub * 16 + (lane & 15);
    int kp = ((lane >> 4) << 4) + t;
    int Kg = ks * 32 + kp;
    wfrag[o] = (_Float16)w[(size_t)M * K_TOT + Kg];
}

// ---------------------------------------------------------------------------
// Main fused kernel: grid = (98 tiles, 8 batches), block = 256 (8 waves).
// ---------------------------------------------------------------------------
__global__ __launch_bounds__(256)
void deform_conv_wmma_kernel(const float* __restrict__ x,
                             const float* __restrict__ off,
                             const v16h*  __restrict__ wfrag,
                             const float* __restrict__ bias,
                             float* __restrict__ out) {
    // B fragments: [ks][nsub][lane][16 halves] ; 36*2*32*16 halves = 73728 B
    __shared__ __align__(32) _Float16 sB[K_STEPS * 2 * 32 * 16];
    __shared__ int   sH0[COMBOS];
    __shared__ int   sW0[COMBOS];
    __shared__ float sWt[COMBOS * 4];

    const int tid = threadIdx.x;
    const int n   = blockIdx.y;
    const int p0  = blockIdx.x * NT;

    // ---- Phase 0: per-(pos,dg,tap) bilinear coords & weights (shared by 64 ch)
    for (int e = tid; e < COMBOS; e += 256) {
        int i   = e / 18;
        int r   = e - i * 18;
        int dg  = r / K_TAPS;
        int tap = r - dg * K_TAPS;
        int khi = tap / 3, kwi = tap - khi * 3;
        int p   = p0 + i;
        int ho  = p / W_IN, wo = p - ho * W_IN;
        size_t obase = ((size_t)n * 36) * P_TOT + p;
        float dh = off[obase + (size_t)((dg * K_TAPS + tap) * 2 + 0) * P_TOT];
        float dw = off[obase + (size_t)((dg * K_TAPS + tap) * 2 + 1) * P_TOT];
        float ph = dh + (float)(khi + ho);   // padded coords (pad = 1)
        float pw = dw + (float)(kwi + wo);
        float h0f = floorf(ph), w0f = floorf(pw);
        float lh = ph - h0f,   lw = pw - w0f;
        sH0[e] = (int)h0f;
        sW0[e] = (int)w0f;
        sWt[e * 4 + 0] = (1.f - lh) * (1.f - lw);
        sWt[e * 4 + 1] = (1.f - lh) * lw;
        sWt[e * 4 + 2] = lh * (1.f - lw);
        sWt[e * 4 + 3] = lh * lw;
    }
    __syncthreads();

    // ---- Phase A: bilinear gather -> LDS in WMMA B-fragment layout.
    // B layout (16-bit, 32x16): N = lane%16, K = (lane>>4)*16 + half.
    for (int j = tid; j < COMBOS * 64; j += 256) {
        int c     = j & 63;
        int combo = j >> 6;
        int i   = combo / 18;
        int r   = combo - i * 18;
        int dg  = r / K_TAPS;
        int tap = r - dg * K_TAPS;

        int h0 = sH0[combo], w0 = sW0[combo];
        float w00 = sWt[combo * 4 + 0], w01 = sWt[combo * 4 + 1];
        float w10 = sWt[combo * 4 + 2], w11 = sWt[combo * 4 + 3];

        // unpadded top-left corner (pad region of padded x is zero)
        int hu = h0 - 1, wu = w0 - 1;
        bool hv0 = (hu >= 0)     & (hu < H_IN);
        bool hv1 = (hu + 1 >= 0) & (hu + 1 < H_IN);
        bool wv0 = (wu >= 0)     & (wu < W_IN);
        bool wv1 = (wu + 1 >= 0) & (wu + 1 < W_IN);

        const float* xc = x + ((size_t)(n * C_IN + dg * 64 + c)) * P_TOT;
        float v = 0.f;
        if (hv0 & wv0) v += w00 * xc[hu * W_IN + wu];
        if (hv0 & wv1) v += w01 * xc[hu * W_IN + wu + 1];
        if (hv1 & wv0) v += w10 * xc[(hu + 1) * W_IN + wu];
        if (hv1 & wv1) v += w11 * xc[(hu + 1) * W_IN + wu + 1];

        int kg   = (dg * 64 + c) * K_TAPS + tap;   // GEMM K row
        int ks   = kg >> 5;
        int kp   = kg & 31;
        int nsub = i >> 4;
        int ncol = i & 15;
        int lane = ((kp >> 4) << 4) + ncol;
        int half = kp & 15;
        sB[(((ks * 2 + nsub) * 32 + lane) << 4) + half] = (_Float16)v;
    }
    __syncthreads();

    // ---- Phase B: GEMM. Wave w owns M-subtiles {2w, 2w+1}, N-subtiles {0,1}.
    const int wave = tid >> 5;
    const int lane = tid & 31;
    const v16h* sBv = reinterpret_cast<const v16h*>(sB);

    v8f acc0 = {}, acc1 = {}, acc2 = {}, acc3 = {};
    const v16h* wa0 = wfrag + ((size_t)(2 * wave)     * K_STEPS) * 32 + lane;
    const v16h* wa1 = wfrag + ((size_t)(2 * wave + 1) * K_STEPS) * 32 + lane;

    for (int ks = 0; ks < K_STEPS; ++ks) {
        v16h a0 = wa0[ks * 32];
        v16h a1 = wa1[ks * 32];
        v16h b0 = sBv[(ks * 2 + 0) * 32 + lane];
        v16h b1 = sBv[(ks * 2 + 1) * 32 + lane];
        acc0 = __builtin_amdgcn_wmma_f32_16x16x32_f16(false, a0, false, b0, (short)0, acc0, false, false);
        acc1 = __builtin_amdgcn_wmma_f32_16x16x32_f16(false, a0, false, b1, (short)0, acc1, false, false);
        acc2 = __builtin_amdgcn_wmma_f32_16x16x32_f16(false, a1, false, b0, (short)0, acc2, false, false);
        acc3 = __builtin_amdgcn_wmma_f32_16x16x32_f16(false, a1, false, b1, (short)0, acc3, false, false);
    }

    // ---- Epilogue: C layout — VGPR r: lanes 0-15 -> M=r, lanes 16-31 -> M=r+8.
    const int ncol  = lane & 15;
    const int rowhi = (lane >> 4) * 8;
    v8f accs[4] = {acc0, acc1, acc2, acc3};
    #pragma unroll
    for (int t = 0; t < 4; ++t) {
        int msub = 2 * wave + (t >> 1);
        int nsub = t & 1;
        int p    = p0 + nsub * 16 + ncol;
        #pragma unroll
        for (int r = 0; r < 8; ++r) {
            int co = msub * 16 + rowhi + r;
            out[((size_t)(n * C_OUT + co)) * P_TOT + p] = accs[t][r] + bias[co];
        }
    }
}

extern "C" void kernel_launch(void* const* d_in, const int* in_sizes, int n_in,
                              void* d_out, int out_size, void* d_ws, size_t ws_size,
                              hipStream_t stream) {
    const float* x    = (const float*)d_in[0];
    const float* off  = (const float*)d_in[1];
    const float* w    = (const float*)d_in[2];
    const float* bias = (const float*)d_in[3];
    float* out        = (float*)d_out;

    _Float16* wfrag = (_Float16*)d_ws;   // 16*36*512 halves = 576 KB

    // Repack weights into WMMA A-fragment order (f32 -> f16).
    int wfrag_elems = 16 * K_STEPS * 512;          // 294912
    repack_weight_kernel<<<wfrag_elems / 256, 256, 0, stream>>>(w, wfrag);

    // Fused deformable conv + WMMA GEMM.
    dim3 grid(N_TILES, N_BATCH);
    deform_conv_wmma_kernel<<<grid, 256, 0, stream>>>(x, off, (const v16h*)wfrag,
                                                      bias, out);
}
